// LSTMAutoEncoder_20375324852354
// MI455X (gfx1250) — matrix operand order repaired and checked by reading, not compiled, not run on hardware
//
#include <hip/hip_runtime.h>
#include <hip/hip_bf16.h>
#include <math.h>

// ---------------------------------------------------------------------------
// LSTM autoencoder (layer-scan, not time-scan) on MI455X / gfx1250.
// bf16 WMMA (v_wmma_f32_16x16x32_bf16), fp32 accumulate.
// Each wave: 4 M-subtiles x 16 cols x 4 gates -> 16 accumulators, so B
// fragments are reused 4x and the load:wmma ratio drops to ~1.0.
// ---------------------------------------------------------------------------

typedef __bf16 bf16_t;
typedef __attribute__((ext_vector_type(16))) __bf16 v16bf;
typedef __attribute__((ext_vector_type(8)))  __bf16 v8bf;
typedef __attribute__((ext_vector_type(8)))  float  v8f;

#define TDIM 4096   // batch rows
#define IDIM 101    // input / decoder hidden dim
#define HDIM 1000   // encoder hidden dim
#define LNUM 4      // layers per stack
#define IP   128    // IDIM padded (mult of 32 for K, 16 for N)
#define HP   1024   // HDIM padded
#define MW   4      // M-subtiles (16 rows each) per wave

// ---------------------------------------------------------------------------
// Packing / init kernels
// ---------------------------------------------------------------------------

__global__ void zero_u32_kernel(unsigned* __restrict__ p, long n) {
    long i = (long)blockIdx.x * blockDim.x + threadIdx.x;
    if (i < n) p[i] = 0u;
}

// x: (T, 1, IDIM) fp32 -> (T, IP) bf16, zero K-padding
__global__ void pack_x_kernel(const float* __restrict__ src, bf16_t* __restrict__ dst) {
    long i = (long)blockIdx.x * blockDim.x + threadIdx.x;
    if (i >= (long)TDIM * IP) return;
    int k = (int)(i % IP);
    long m = i / IP;
    float v = (k < IDIM) ? src[m * IDIM + k] : 0.f;
    dst[i] = (bf16_t)v;
}

// src: one layer's weight (4*Ng rows, Kv cols) fp32 row-major
// dst: (4, Npad, Kpad) bf16, zero padded in both N (rows/gate) and K.
__global__ void pack_w_kernel(const float* __restrict__ src, bf16_t* __restrict__ dst,
                              int Ng, int Kv, int Npad, int Kpad) {
    long i = (long)blockIdx.x * blockDim.x + threadIdx.x;
    long total = (long)4 * Npad * Kpad;
    if (i >= total) return;
    int k = (int)(i % Kpad);
    int n = (int)((i / Kpad) % Npad);
    int g = (int)(i / ((long)Kpad * Npad));
    float v = 0.f;
    if (n < Ng && k < Kv) v = src[(long)(g * Ng + n) * Kv + k];
    dst[i] = (bf16_t)v;
}

// combined bias: dst[g*Npad+n] = bih[g*Ng+n] + bhh[g*Ng+n], zero padded
__global__ void pack_b_kernel(const float* __restrict__ bih, const float* __restrict__ bhh,
                              float* __restrict__ dst, int Ng, int Npad) {
    int i = blockIdx.x * blockDim.x + threadIdx.x;
    if (i >= 4 * Npad) return;
    int n = i % Npad;
    int g = i / Npad;
    float v = 0.f;
    if (n < Ng) v = bih[g * Ng + n] + bhh[g * Ng + n];
    dst[i] = v;
}

// ---------------------------------------------------------------------------
// Fused WMMA layer kernel
// ---------------------------------------------------------------------------

// acc[mi][g] += A(m0+16*mi .. +16, :) x W[g](n0..n0+16, :)^T
// A: (TDIM, K) bf16 row-major (K mult of 32).  W: (4, Npad, K) bf16.
__device__ __forceinline__ void gemm_mw_acc(const bf16_t* __restrict__ A,
                                            const bf16_t* __restrict__ W,
                                            int K, int Npad, int m0, int n0, int lane,
                                            v8f acc[MW][4]) {
    const int ar = lane & 15;            // A row within subtile
    const int ak = (lane >> 4) << 4;     // A K sub-offset (0 or 16)
    const int bn = lane & 15;            // B column (= W row within gate)
    const int bk = (lane >> 4) << 4;     // B K sub-offset (0 or 16)
    const bf16_t* ap = A + (long)(m0 + ar) * K + ak;
    const bf16_t* bp = W + (long)(n0 + bn) * K + bk;
    const long gs = (long)Npad * K;      // gate stride in W
    const long ms = (long)16 * K;        // M-subtile stride in A

    for (int k0 = 0; k0 < K; k0 += 32) {
        // B fragments: contiguous 16 K-values of column bn per lane, one per gate
        v16bf b[4];
#pragma unroll
        for (int g = 0; g < 4; ++g) b[g] = *(const v16bf*)(bp + g * gs + k0);

        __builtin_prefetch(bp + k0 + 64, 0, 0);            // global_prefetch_b8
        __builtin_prefetch(bp + 3 * gs + k0 + 64, 0, 0);

#pragma unroll
        for (int mi = 0; mi < MW; ++mi) {
            // A fragment: lanes 0-15 K(k0+0..15), lanes 16-31 K(k0+16..31)
            v8bf alo = *(const v8bf*)(ap + mi * ms + k0);
            v8bf ahi = *(const v8bf*)(ap + mi * ms + k0 + 8);
            v16bf a;
#pragma unroll
            for (int i = 0; i < 8; ++i) { a[i] = alo[i]; a[i + 8] = ahi[i]; }
#pragma unroll
            for (int g = 0; g < 4; ++g)
                acc[mi][g] = __builtin_amdgcn_wmma_f32_16x16x32_bf16(
                    false, a, false, b[g], (short)0, acc[mi][g], false, false);
        }
    }
}

__device__ __forceinline__ float sigm(float x) { return 1.f / (1.f + __expf(-x)); }

// gates = A1 @ W1^T + A2 @ W2^T + bias; c = sig(f)*c + sig(i)*tanh(g); h = sig(o)*tanh(c)
__global__ __launch_bounds__(128) void lstm_layer_wmma(
        const bf16_t* __restrict__ A1, const bf16_t* __restrict__ W1, int K1,
        const bf16_t* __restrict__ A2, const bf16_t* __restrict__ W2, int K2,
        const float* __restrict__ bias,
        float* __restrict__ C,             // (TDIM, Npad) fp32 cell state, in/out
        bf16_t* __restrict__ Hb,           // (TDIM, Npad) bf16 h output
        float* __restrict__ Hf,            // (TDIM, Npad) fp32 h output (when do_f32)
        int Npad, int do_relu, int do_f32) {
    const int lane = threadIdx.x & 31;
    const int wid = blockIdx.x * (blockDim.x >> 5) + (threadIdx.x >> 5);
    const int ntiles = Npad >> 4;
    const int nt = wid % ntiles;
    const int mt = wid / ntiles;
    const int m0 = mt * (16 * MW);
    const int n0 = nt << 4;

    v8f acc[MW][4];
#pragma unroll
    for (int mi = 0; mi < MW; ++mi)
#pragma unroll
        for (int g = 0; g < 4; ++g) acc[mi][g] = (v8f){};

    gemm_mw_acc(A1, W1, K1, Npad, m0, n0, lane, acc);
    gemm_mw_acc(A2, W2, K2, Npad, m0, n0, lane, acc);

    // C/D layout: lane holds column n0+(lane&15); element e holds row base + 8*(lane>>4) + e
    const int col = n0 + (lane & 15);
    const float bI = bias[col];
    const float bF = bias[Npad + col];
    const float bG = bias[2 * Npad + col];
    const float bO = bias[3 * Npad + col];

#pragma unroll
    for (int mi = 0; mi < MW; ++mi) {
        const int rb = m0 + mi * 16 + ((lane >> 4) << 3);
#pragma unroll
        for (int e = 0; e < 8; ++e) {
            long off = (long)(rb + e) * Npad + col;
            float gi = acc[mi][0][e] + bI;
            float gf = acc[mi][1][e] + bF;
            float gg = acc[mi][2][e] + bG;
            float go = acc[mi][3][e] + bO;
            float cp = C[off];
            float c = sigm(gf) * cp + sigm(gi) * tanhf(gg);
            float h = sigm(go) * tanhf(c);
            if (do_relu) h = fmaxf(h, 0.f);
            C[off] = c;
            Hb[off] = (bf16_t)h;
            if (do_f32) Hf[off] = h;
        }
    }
}

// ---------------------------------------------------------------------------
// log_softmax over the valid 101 columns; one wave per row
// ---------------------------------------------------------------------------
__global__ __launch_bounds__(128) void logsoftmax_kernel(const float* __restrict__ Hf,
                                                         float* __restrict__ out) {
    int row = blockIdx.x * (blockDim.x >> 5) + (threadIdx.x >> 5);
    if (row >= TDIM) return;
    int lane = threadIdx.x & 31;
    const float* p = Hf + (long)row * IP;

    float mx = -3.0e38f;
    for (int j = lane; j < IDIM; j += 32) mx = fmaxf(mx, p[j]);
#pragma unroll
    for (int o = 16; o > 0; o >>= 1) mx = fmaxf(mx, __shfl_xor(mx, o, 32));

    float s = 0.f;
    for (int j = lane; j < IDIM; j += 32) s += __expf(p[j] - mx);
#pragma unroll
    for (int o = 16; o > 0; o >>= 1) s += __shfl_xor(s, o, 32);

    float lse = mx + __logf(s);
    for (int j = lane; j < IDIM; j += 32) out[(long)row * IDIM + j] = p[j] - lse;
}

// ---------------------------------------------------------------------------
// Host launch
// ---------------------------------------------------------------------------
extern "C" void kernel_launch(void* const* d_in, const int* in_sizes, int n_in,
                              void* d_out, int out_size, void* d_ws, size_t ws_size,
                              hipStream_t stream) {
    const float* x    = (const float*)d_in[0];
    const float* eWih = (const float*)d_in[1];
    const float* eWhh = (const float*)d_in[2];
    const float* ebih = (const float*)d_in[3];
    const float* ebhh = (const float*)d_in[4];
    const float* dWih = (const float*)d_in[5];
    const float* dWhh = (const float*)d_in[6];
    const float* dbih = (const float*)d_in[7];
    const float* dbhh = (const float*)d_in[8];
    float* out = (float*)d_out;

    char* ws = (char*)d_ws;
    size_t off = 0;
    auto carve = [&](size_t bytes) -> char* {
        char* p = ws + off;
        off += (bytes + 255) & ~(size_t)255;
        return p;
    };

    bf16_t* X    = (bf16_t*)carve((size_t)TDIM * IP * 2);   // padded bf16 x
    bf16_t* H0   = (bf16_t*)carve((size_t)TDIM * HP * 2);   // encoder h ping
    bf16_t* H1   = (bf16_t*)carve((size_t)TDIM * HP * 2);   // encoder h pong
    bf16_t* ENC  = (bf16_t*)carve((size_t)TDIM * HP * 2);   // relu(enc h), decoder input
    float*  CE   = (float*) carve((size_t)TDIM * HP * 4);   // encoder cell state
    bf16_t* HD0  = (bf16_t*)carve((size_t)TDIM * IP * 2);   // decoder h ping
    bf16_t* HD1  = (bf16_t*)carve((size_t)TDIM * IP * 2);   // decoder h pong
    float*  CD   = (float*) carve((size_t)TDIM * IP * 4);   // decoder cell state
    float*  HDF  = (float*) carve((size_t)TDIM * IP * 4);   // final decoder h (fp32)
    bf16_t* WIH  = (bf16_t*)carve((size_t)4 * HP * IP * 2); // per-layer packed Wih
    bf16_t* WHH  = (bf16_t*)carve((size_t)4 * HP * HP * 2); // per-layer packed Whh (max)
    float*  BB   = (float*) carve((size_t)4 * HP * 4);      // per-layer packed bias

    const int ZT = 256;
    // zero encoder state (h=0, c=0)
    {
        long nh = (long)TDIM * HP / 2;  // bf16 words
        long nc = (long)TDIM * HP;      // fp32 words
        zero_u32_kernel<<<(unsigned)((nh + ZT - 1) / ZT), ZT, 0, stream>>>((unsigned*)H0, nh);
        zero_u32_kernel<<<(unsigned)((nc + ZT - 1) / ZT), ZT, 0, stream>>>((unsigned*)CE, nc);
    }
    // pack x
    {
        long n = (long)TDIM * IP;
        pack_x_kernel<<<(unsigned)((n + ZT - 1) / ZT), ZT, 0, stream>>>(x, X);
    }

    // -------- encoder stack: M=4096, N/gate=1000->HP, K1=101->IP, K2=1000->HP
    const unsigned encBlocks = (unsigned)(((TDIM / (16 * MW)) * (HP / 16)) / 4); // 4 waves/block
    for (int l = 0; l < LNUM; ++l) {
        long nw;
        nw = (long)4 * HP * IP;
        pack_w_kernel<<<(unsigned)((nw + ZT - 1) / ZT), ZT, 0, stream>>>(
            eWih + (long)l * 4 * HDIM * IDIM, WIH, HDIM, IDIM, HP, IP);
        nw = (long)4 * HP * HP;
        pack_w_kernel<<<(unsigned)((nw + ZT - 1) / ZT), ZT, 0, stream>>>(
            eWhh + (long)l * 4 * HDIM * HDIM, WHH, HDIM, HDIM, HP, HP);
        pack_b_kernel<<<(4 * HP + ZT - 1) / ZT, ZT, 0, stream>>>(
            ebih + (long)l * 4 * HDIM, ebhh + (long)l * 4 * HDIM, BB, HDIM, HP);

        const bf16_t* hin = (l % 2 == 0) ? H0 : H1;
        bf16_t* hout = (l == 3) ? ENC : ((l % 2 == 0) ? H1 : H0);
        lstm_layer_wmma<<<encBlocks, 128, 0, stream>>>(
            X, WIH, IP, hin, WHH, HP, BB, CE, hout, HDF, HP,
            /*do_relu=*/(l == 3) ? 1 : 0, /*do_f32=*/0);
    }

    // zero decoder state
    {
        long nh = (long)TDIM * IP / 2;
        long nc = (long)TDIM * IP;
        zero_u32_kernel<<<(unsigned)((nh + ZT - 1) / ZT), ZT, 0, stream>>>((unsigned*)HD0, nh);
        zero_u32_kernel<<<(unsigned)((nc + ZT - 1) / ZT), ZT, 0, stream>>>((unsigned*)CD, nc);
    }

    // -------- decoder stack: M=4096, N/gate=101->IP, K1=1000->HP, K2=101->IP
    const unsigned decBlocks = (unsigned)(((TDIM / (16 * MW)) * (IP / 16)) / 4);
    for (int l = 0; l < LNUM; ++l) {
        long nw;
        nw = (long)4 * IP * HP;
        pack_w_kernel<<<(unsigned)((nw + ZT - 1) / ZT), ZT, 0, stream>>>(
            dWih + (long)l * 4 * IDIM * HDIM, WIH, IDIM, HDIM, IP, HP);
        nw = (long)4 * IP * IP;
        pack_w_kernel<<<(unsigned)((nw + ZT - 1) / ZT), ZT, 0, stream>>>(
            dWhh + (long)l * 4 * IDIM * IDIM, WHH, IDIM, IDIM, IP, IP);
        pack_b_kernel<<<(4 * IP + ZT - 1) / ZT, ZT, 0, stream>>>(
            dbih + (long)l * 4 * IDIM, dbhh + (long)l * 4 * IDIM, BB, IDIM, IP);

        const bf16_t* hin = (l % 2 == 0) ? HD0 : HD1;
        bf16_t* hout = (l % 2 == 0) ? HD1 : HD0;
        lstm_layer_wmma<<<decBlocks, 128, 0, stream>>>(
            ENC, WIH, HP, hin, WHH, IP, BB, CD, hout, HDF, IP,
            /*do_relu=*/0, /*do_f32=*/(l == 3) ? 1 : 0);
    }

    // log_softmax over valid 101 columns -> output (T, 1, IDIM)
    logsoftmax_kernel<<<TDIM / 4, 128, 0, stream>>>(HDF, out);
}
